// KalmanNetLatentNN_52243982188765
// MI455X (gfx1250) — matrix operand order repaired and checked by reading, not compiled, not run on hardware
//
#include <hip/hip_runtime.h>

// ============================================================================
// KalmanNet recurrent step on MI455X (gfx1250).
// Memory-bound: ~557MB fp32 weights would be re-streamed each of 128 steps.
// Strategy: one-time fp32->bf16 weight conversion into d_ws (halves HBM
// traffic), then all big GEMMs via v_wmma_f32_16x16x32_bf16 (f32 accumulate).
// Wave32: each wave owns a 32x16 output slab (two 16x16 WMMA tiles sharing one
// B fragment) over a 1/4 K-slice; 4 K-slices reduce through LDS. Grid = Nout/32
// blocks (e.g. 216 blocks / 1728 waves for the 6912-wide GRU gate GEMMs) to put
// enough b128 loads in flight to approach the 23.3 TB/s HBM roof.
// Workspace requirement: ~290 MB.
// ============================================================================

typedef __bf16 bf16_t;
typedef __attribute__((ext_vector_type(16))) __bf16 v16bf;
typedef __attribute__((ext_vector_type(8)))  __bf16 v8bf;
typedef __attribute__((ext_vector_type(8)))  float  v8f;

// ---------------------------------------------------------------------------
// fp32 -> bf16 weight conversion with K padding (pad columns zeroed)
// ---------------------------------------------------------------------------
__global__ void k_convert(const float* __restrict__ src, bf16_t* __restrict__ dst,
                          int R, int Kin, int Kpad) {
  int idx = blockIdx.x * blockDim.x + threadIdx.x;
  int total = R * Kpad;
  if (idx >= total) return;
  int r = idx / Kpad;
  int c = idx - r * Kpad;
  float v = (c < Kin) ? src[(size_t)r * Kin + c] : 0.0f;
  dst[idx] = (bf16_t)v;
}

__global__ void k_zero_bf16(bf16_t* __restrict__ p, int count) {
  int idx = blockIdx.x * blockDim.x + threadIdx.x;
  if (idx < count) p[idx] = (bf16_t)0.0f;
}

// copy fp32 src[rows,cols] into optional f32 dst (ldF) and/or bf16 dst (ldB)
__global__ void k_copy_cast(const float* __restrict__ src, int rows, int cols,
                            float* __restrict__ dstF, int ldF,
                            bf16_t* __restrict__ dstB, int ldB) {
  int idx = blockIdx.x * blockDim.x + threadIdx.x;
  if (idx >= rows * cols) return;
  int rr = idx / cols;
  int cc = idx - rr * cols;
  float v = src[idx];
  if (dstF) dstF[(size_t)rr * ldF + cc] = v;
  if (dstB) dstB[(size_t)rr * ldB + cc] = (bf16_t)v;
}

// ---------------------------------------------------------------------------
// WMMA GEMM:  out[32, Nout] = act( A_bf16[32,K](lda) @ W_bf16[Nout,K]^T + bias )
// Block = 256 threads = 8 waves = 2 column tiles x 4 K-slices. Each wave runs
// the bf16 WMMA pipeline over its K-slice (chunks of 32); wave index is forced
// into an SGPR via readfirstlane so slice bounds / column base are scalar
// (scalar loop control, 4x unroll, deep load clauses). Partial accumulators
// are reduced through LDS (17-float padded rows -> conflict-free), and waves
// 0/1 apply bias/activation and write the f32/bf16 outputs.
// A-frag per ISA 16-bit 16x32 layout: lane l (r=l&15, half=l>>4) holds
// K = k0+half*8 .. +7 (elems 0-7) and K = k0+16+half*8 .. +7 (elems 8-15).
// B-frag (32x16, K x N): lane column n=r, same K order as A-frag -> two
// 16B loads from row-major W (since B[k][n] == W[n][k]).
// ---------------------------------------------------------------------------
__global__ __launch_bounds__(256) void k_gemm(
    const bf16_t* __restrict__ A, int lda,
    const bf16_t* __restrict__ W, int K,
    const float* __restrict__ bias,
    float* __restrict__ outF, int ldF,
    bf16_t* __restrict__ outB1, int ldB1,
    bf16_t* __restrict__ outB2, int ldB2,
    int relu) {
  // wave id is uniform across the wave: pin it to an SGPR so the K-slice loop
  // gets scalar control flow instead of an exec-masked loop.
  const int wave    = __builtin_amdgcn_readfirstlane(threadIdx.x >> 5);
  const int lane    = threadIdx.x & 31;
  const int half    = lane >> 4;
  const int r       = lane & 15;
  const int colTile = wave & 1;   // 2 column tiles per block
  const int kslice  = wave >> 1;  // 4 K-slices per column tile
  const int col0    = (blockIdx.x * 2 + colTile) * 16;

  // Telescoping K-chunk partition: exact cover for any K % 32 == 0.
  const int nk    = K >> 5;
  const int c_beg = (nk * kslice) >> 2;
  const int c_end = (nk * (kslice + 1)) >> 2;

  const bf16_t* __restrict__ a0p = A + (size_t)r        * lda + half * 8;
  const bf16_t* __restrict__ a1p = A + (size_t)(r + 16) * lda + half * 8;
  const bf16_t* __restrict__ bp  = W + (size_t)(col0 + r) * K + half * 8;

  v8f c0 = {};
  v8f c1 = {};
#pragma unroll 4
  for (int ck = c_beg; ck < c_end; ++ck) {
    const int k0 = ck << 5;
    union { v16bf v; v8bf h[2]; } a0, a1, bb;
    a0.h[0] = *(const v8bf*)(a0p + k0);
    a0.h[1] = *(const v8bf*)(a0p + k0 + 16);
    a1.h[0] = *(const v8bf*)(a1p + k0);
    a1.h[1] = *(const v8bf*)(a1p + k0 + 16);
    bb.h[0] = *(const v8bf*)(bp + k0);
    bb.h[1] = *(const v8bf*)(bp + k0 + 16);
    __builtin_prefetch(bp + k0 + 512, 0, 1);   // global_prefetch_b8 on W stream
    c0 = __builtin_amdgcn_wmma_f32_16x16x32_bf16(false, a0.v, false, bb.v,
                                                 (short)0, c0, false, false);
    c1 = __builtin_amdgcn_wmma_f32_16x16x32_bf16(false, a1.v, false, bb.v,
                                                 (short)0, c1, false, false);
  }

  // ---- cross-wave K-slice reduction through LDS ----
  __shared__ float red[8][32][17];   // pad 16->17 floats: conflict-free
#pragma unroll
  for (int i = 0; i < 8; ++i) {
    red[wave][lane][i]     = c0[i];
    red[wave][lane][8 + i] = c1[i];
  }
  __syncthreads();
  if (wave < 2) {   // scalar branch (wave is SGPR)
    // this wave holds slice 0 of column tile `wave`; add slices 1..3
#pragma unroll
    for (int s = 1; s < 4; ++s) {
      const int src = wave + s * 2;
#pragma unroll
      for (int i = 0; i < 8; ++i) {
        c0[i] += red[src][lane][i];
        c1[i] += red[src][lane][8 + i];
      }
    }
    // C/D layout: lane l -> col n = l&15; VGPR i -> row = half*8 + i (tile 0),
    // +16 for tile 1.
    const int n = col0 + r;
    const float bcol = bias ? bias[n] : 0.0f;
#pragma unroll
    for (int i = 0; i < 8; ++i) {
      int row0 = half * 8 + i;
      int row1 = row0 + 16;
      float v0 = c0[i] + bcol;
      float v1 = c1[i] + bcol;
      if (relu) { v0 = fmaxf(v0, 0.0f); v1 = fmaxf(v1, 0.0f); }
      if (outF) {
        outF[(size_t)row0 * ldF + n] = v0;
        outF[(size_t)row1 * ldF + n] = v1;
      }
      if (outB1) {
        outB1[(size_t)row0 * ldB1 + n] = (bf16_t)v0;
        outB1[(size_t)row1 * ldB1 + n] = (bf16_t)v1;
      }
      if (outB2) {
        outB2[(size_t)row0 * ldB2 + n] = (bf16_t)v0;
        outB2[(size_t)row1 * ldB2 + n] = (bf16_t)v1;
      }
    }
  }
}

// ---------------------------------------------------------------------------
// GRU gate combine: gi/gh are [32,6912] f32 (bias already added by GEMM).
// h' = (1-z)*tanh(in + r*hn) + z*h ; writes f32 carry and up to two bf16 views.
// ---------------------------------------------------------------------------
__device__ __forceinline__ float sigmoidf_(float x) {
  return 1.0f / (1.0f + __expf(-x));
}

__global__ void k_gru_combine(const float* __restrict__ gi, const float* __restrict__ gh,
                              const float* __restrict__ h,
                              float* __restrict__ outF,
                              bf16_t* __restrict__ outB1, int ldB1,
                              bf16_t* __restrict__ outB2, int ldB2) {
  int idx = blockIdx.x * blockDim.x + threadIdx.x;
  if (idx >= 32 * 2304) return;
  int b = idx / 2304;
  int j = idx - b * 2304;
  const float* gib = gi + (size_t)b * 6912;
  const float* ghb = gh + (size_t)b * 6912;
  float ir = gib[j], iz = gib[j + 2304], in = gib[j + 4608];
  float hr = ghb[j], hz = ghb[j + 2304], hn = ghb[j + 4608];
  float rg = sigmoidf_(ir + hr);
  float zg = sigmoidf_(iz + hz);
  float ng = tanhf(in + rg * hn);
  float hnew = (1.0f - zg) * ng + zg * h[idx];
  if (outF)  outF[idx] = hnew;
  if (outB1) outB1[(size_t)b * ldB1 + j] = (bf16_t)hnew;
  if (outB2) outB2[(size_t)b * ldB2 + j] = (bf16_t)hnew;
}

// ---------------------------------------------------------------------------
// Small per-step prologue (one block per batch row):
// x_prior = F@x ; m1y = H@x_prior ; four l2-normalized features ;
// o5=relu(W5@fe+b5) -> xq (bf16) ; o6 -> catQ6[:,2304:] ; o7 -> catS_in[:,2304:]
// dy = y - m1y (f32) ; x_prior (f32).
// ---------------------------------------------------------------------------
__global__ void k_small_pre(const float* __restrict__ y_t,
                            const float* __restrict__ F, const float* __restrict__ H,
                            const float* __restrict__ W5, const float* __restrict__ b5,
                            const float* __restrict__ W6, const float* __restrict__ b6,
                            const float* __restrict__ W7, const float* __restrict__ b7,
                            const float* __restrict__ x_post, const float* __restrict__ x_post_prev,
                            const float* __restrict__ x_prior_prev, const float* __restrict__ y_prev,
                            float* __restrict__ x_prior, float* __restrict__ dy,
                            bf16_t* __restrict__ xq, bf16_t* __restrict__ cQ6_o6,
                            bf16_t* __restrict__ cS_o7) {
  const int b = blockIdx.x;
  const int t = threadIdx.x;  // blockDim = 128
  __shared__ float xs[48], xp[48], sy[48], syp[48], sxpp[48], sxrp[48];
  __shared__ float dod[48], doi[48], dfe[48], dfu[48];
  __shared__ float nrm[4];
  if (t < 48) {
    xs[t]   = x_post[b * 48 + t];
    sy[t]   = y_t[b * 48 + t];
    syp[t]  = y_prev[b * 48 + t];
    sxpp[t] = x_post_prev[b * 48 + t];
    sxrp[t] = x_prior_prev[b * 48 + t];
  }
  __syncthreads();
  if (t < 48) {
    float acc = 0.0f;
    for (int j = 0; j < 48; ++j) acc += F[t * 48 + j] * xs[j];
    xp[t] = acc;
    x_prior[b * 48 + t] = acc;
  }
  __syncthreads();
  if (t < 48) {
    float acc = 0.0f;
    for (int j = 0; j < 48; ++j) acc += H[t * 48 + j] * xp[j];
    float d = sy[t] - acc;       // y - m1y
    doi[t] = d;
    dy[b * 48 + t] = d;
    dod[t] = sy[t] - syp[t];
    dfe[t] = xs[t] - sxpp[t];
    dfu[t] = xs[t] - sxrp[t];
  }
  __syncthreads();
  if (t < 4) {
    const float* v = (t == 0) ? dod : (t == 1) ? doi : (t == 2) ? dfe : dfu;
    float s = 0.0f;
    for (int j = 0; j < 48; ++j) s += v[j] * v[j];
    nrm[t] = fmaxf(sqrtf(s), 1e-12f);
  }
  __syncthreads();
  if (t < 48) {
    dod[t] /= nrm[0]; doi[t] /= nrm[1]; dfe[t] /= nrm[2]; dfu[t] /= nrm[3];
  }
  __syncthreads();
  if (t < 48) {
    float a5 = b5[t], a6 = b6[t];
    for (int j = 0; j < 48; ++j) {
      a5 += W5[t * 48 + j] * dfe[j];
      a6 += W6[t * 48 + j] * dfu[j];
    }
    xq[b * 64 + t]        = (bf16_t)fmaxf(a5, 0.0f);
    cQ6_o6[b * 2368 + t]  = (bf16_t)fmaxf(a6, 0.0f);
  }
  if (t < 96) {
    float a7 = b7[t];
    for (int j = 0; j < 96; ++j) {
      float inj = (j < 48) ? dod[j] : doi[j - 48];
      a7 += W7[t * 96 + j] * inj;
    }
    cS_o7[b * 2400 + t] = (bf16_t)fmaxf(a7, 0.0f);
  }
}

// ---------------------------------------------------------------------------
// Small per-step epilogue: x_new = x_prior + KG@dy (KG = o2 reshaped 48x48),
// write output slice, shift carries.
// ---------------------------------------------------------------------------
__global__ void k_small_post(const float* __restrict__ o2f, const float* __restrict__ x_prior,
                             const float* __restrict__ dy, const float* __restrict__ y_t,
                             float* __restrict__ x_post, float* __restrict__ x_post_prev,
                             float* __restrict__ x_prior_prev, float* __restrict__ y_prev,
                             float* __restrict__ out_t) {
  const int b = blockIdx.x;
  const int t = threadIdx.x;  // blockDim = 64
  __shared__ float dys[48];
  if (t < 48) dys[t] = dy[b * 48 + t];
  __syncthreads();
  if (t < 48) {
    float acc = x_prior[b * 48 + t];
    const float* kg = o2f + (size_t)b * 2304 + t * 48;
    for (int j = 0; j < 48; ++j) acc += kg[j] * dys[j];
    float oldx = x_post[b * 48 + t];
    out_t[b * 48 + t]        = acc;
    x_post_prev[b * 48 + t]  = oldx;
    x_post[b * 48 + t]       = acc;
    x_prior_prev[b * 48 + t] = x_prior[b * 48 + t];
    y_prev[b * 48 + t]       = y_t[b * 48 + t];
  }
}

__global__ void k_init_yprev(const float* __restrict__ x0, const float* __restrict__ H,
                             float* __restrict__ y_prev) {
  const int b = blockIdx.x;
  const int t = threadIdx.x;  // blockDim = 64
  __shared__ float xs[48];
  if (t < 48) xs[t] = x0[b * 48 + t];
  __syncthreads();
  if (t < 48) {
    float acc = 0.0f;
    for (int j = 0; j < 48; ++j) acc += H[t * 48 + j] * xs[j];
    y_prev[b * 48 + t] = acc;
  }
}

// ===========================================================================
extern "C" void kernel_launch(void* const* d_in, const int* in_sizes, int n_in,
                              void* d_out, int out_size, void* d_ws, size_t ws_size,
                              hipStream_t stream) {
  (void)in_sizes; (void)n_in; (void)out_size; (void)ws_size;

  const float* y_seq = (const float*)d_in[0];
  const float* x0    = (const float*)d_in[1];
  const float* hQ0   = (const float*)d_in[2];
  const float* hSig0 = (const float*)d_in[3];
  const float* hS0   = (const float*)d_in[4];
  const float* F     = (const float*)d_in[5];
  const float* Hm    = (const float*)d_in[6];
  const float* WiQ   = (const float*)d_in[7];
  const float* WhQ   = (const float*)d_in[8];
  const float* biQ   = (const float*)d_in[9];
  const float* bhQ   = (const float*)d_in[10];
  const float* WiSig = (const float*)d_in[11];
  const float* WhSig = (const float*)d_in[12];
  const float* biSig = (const float*)d_in[13];
  const float* bhSig = (const float*)d_in[14];
  const float* WiS   = (const float*)d_in[15];
  const float* WhS   = (const float*)d_in[16];
  const float* biS   = (const float*)d_in[17];
  const float* bhS   = (const float*)d_in[18];
  const float* W1    = (const float*)d_in[19];
  const float* b1    = (const float*)d_in[20];
  const float* W2a   = (const float*)d_in[21];
  const float* b2a   = (const float*)d_in[22];
  const float* W2b   = (const float*)d_in[23];
  const float* b2b   = (const float*)d_in[24];
  const float* W3    = (const float*)d_in[25];
  const float* b3    = (const float*)d_in[26];
  const float* W4    = (const float*)d_in[27];
  const float* b4    = (const float*)d_in[28];
  const float* W5    = (const float*)d_in[29];
  const float* b5    = (const float*)d_in[30];
  const float* W6    = (const float*)d_in[31];
  const float* b6    = (const float*)d_in[32];
  const float* W7    = (const float*)d_in[33];
  const float* b7    = (const float*)d_in[34];
  float* out = (float*)d_out;

  // ---- workspace carve-up (256B aligned blocks) ----
  char* ws = (char*)d_ws;
  size_t off = 0;
  auto take = [&](size_t elems, size_t esize) -> char* {
    char* p = ws + off;
    off += (elems * esize + 255) & ~(size_t)255;
    return p;
  };
  // bf16 weights (padded K where needed)
  bf16_t* WiQb   = (bf16_t*)take((size_t)6912 * 64,   2);
  bf16_t* WhQb   = (bf16_t*)take((size_t)6912 * 2304, 2);
  bf16_t* WiSigb = (bf16_t*)take((size_t)6912 * 2368, 2);
  bf16_t* WhSigb = (bf16_t*)take((size_t)6912 * 2304, 2);
  bf16_t* WiSb   = (bf16_t*)take((size_t)6912 * 2400, 2);
  bf16_t* WhSb   = (bf16_t*)take((size_t)6912 * 2304, 2);
  bf16_t* W1b    = (bf16_t*)take((size_t)2304 * 2304, 2);
  bf16_t* W2ab   = (bf16_t*)take((size_t)4608 * 4608, 2);
  bf16_t* W2bb   = (bf16_t*)take((size_t)2304 * 4608, 2);
  bf16_t* W3b    = (bf16_t*)take((size_t)2304 * 4608, 2);
  bf16_t* W4b    = (bf16_t*)take((size_t)2304 * 4608, 2);
  // bf16 activations (concat slabs double as GRU-h inputs for the next step)
  bf16_t* xq      = (bf16_t*)take(32 * 64,   2);  // o5, K padded 48->64
  bf16_t* catQ6   = (bf16_t*)take(32 * 2368, 2);  // [hQ | o6 | pad16]
  bf16_t* hSigb   = (bf16_t*)take(32 * 2304, 2);  // hSigma carry (bf16)
  bf16_t* catSin  = (bf16_t*)take(32 * 2400, 2);  // [o1 | o7]
  bf16_t* catSigS = (bf16_t*)take(32 * 4608, 2);  // [hSig_n | hS_n]
  bf16_t* o2ab    = (bf16_t*)take(32 * 4608, 2);  // relu(W2a..)
  bf16_t* catSo2  = (bf16_t*)take(32 * 4608, 2);  // [hS_n | o2]
  bf16_t* catSg3  = (bf16_t*)take(32 * 4608, 2);  // [hSig_n | o3]
  // f32 buffers
  float* gi      = (float*)take((size_t)32 * 6912, 4);
  float* gh      = (float*)take((size_t)32 * 6912, 4);
  float* hQf     = (float*)take(32 * 2304, 4);
  float* hSigf   = (float*)take(32 * 2304, 4);
  float* hSf     = (float*)take(32 * 2304, 4);
  float* o2f     = (float*)take(32 * 2304, 4);
  float* xpost   = (float*)take(32 * 48, 4);
  float* xpostp  = (float*)take(32 * 48, 4);
  float* xpriorp = (float*)take(32 * 48, 4);
  float* xprior  = (float*)take(32 * 48, 4);
  float* yprev   = (float*)take(32 * 48, 4);
  float* dyb     = (float*)take(32 * 48, 4);

  auto convert = [&](const float* src, bf16_t* dst, int R, int Kin, int Kpad) {
    int total = R * Kpad;
    k_convert<<<(total + 255) / 256, 256, 0, stream>>>(src, dst, R, Kin, Kpad);
  };
  auto gemm = [&](const bf16_t* A, int lda, const bf16_t* Wb, int K,
                  const float* bias, float* oF, int ldF,
                  bf16_t* oB1, int ldB1, bf16_t* oB2, int ldB2,
                  int relu, int Nout) {
    k_gemm<<<Nout / 32, 256, 0, stream>>>(A, lda, Wb, K, bias, oF, ldF,
                                          oB1, ldB1, oB2, ldB2, relu);
  };
  auto copycast = [&](const float* src, int rows, int cols, float* dF, int ldF,
                      bf16_t* dB, int ldB) {
    k_copy_cast<<<(rows * cols + 255) / 256, 256, 0, stream>>>(src, rows, cols,
                                                               dF, ldF, dB, ldB);
  };

  // ---- one-time (per call) weight conversion to bf16 ----
  convert(WiQ,   WiQb,   6912, 48,   64);
  convert(WhQ,   WhQb,   6912, 2304, 2304);
  convert(WiSig, WiSigb, 6912, 2352, 2368);
  convert(WhSig, WhSigb, 6912, 2304, 2304);
  convert(WiS,   WiSb,   6912, 2400, 2400);
  convert(WhS,   WhSb,   6912, 2304, 2304);
  convert(W1,    W1b,    2304, 2304, 2304);
  convert(W2a,   W2ab,   4608, 4608, 4608);
  convert(W2b,   W2bb,   2304, 4608, 4608);
  convert(W3,    W3b,    2304, 4608, 4608);
  convert(W4,    W4b,    2304, 4608, 4608);

  // ---- state init (pads zeroed, carries seeded) ----
  k_zero_bf16<<<(32 * 64 + 255) / 256, 256, 0, stream>>>(xq, 32 * 64);
  k_zero_bf16<<<(32 * 2368 + 255) / 256, 256, 0, stream>>>(catQ6, 32 * 2368);
  copycast(x0, 32, 48, xpost,   48, nullptr, 0);
  copycast(x0, 32, 48, xpostp,  48, nullptr, 0);
  copycast(x0, 32, 48, xpriorp, 48, nullptr, 0);
  copycast(hQ0,   32, 2304, hQf,   2304, catQ6,  2368);  // hQ lives in catQ6[:, :2304]
  copycast(hSig0, 32, 2304, hSigf, 2304, hSigb,  2304);
  copycast(hS0,   32, 2304, hSf,   2304, catSo2, 4608);  // hS lives in catSo2[:, :2304]
  k_init_yprev<<<32, 64, 0, stream>>>(x0, Hm, yprev);

  // ---- time loop: stream-ordered stage launches (graph-capture friendly) ----
  for (int t = 0; t < 128; ++t) {
    const float* yt = y_seq + (size_t)t * 32 * 48;

    k_small_pre<<<32, 128, 0, stream>>>(yt, F, Hm, W5, b5, W6, b6, W7, b7,
                                        xpost, xpostp, xpriorp, yprev,
                                        xprior, dyb, xq, catQ6 + 2304, catSin + 2304);

    // GRU_Q
    gemm(xq,    64,   WiQb, 64,   biQ, gi, 6912, nullptr, 0, nullptr, 0, 0, 6912);
    gemm(catQ6, 2368, WhQb, 2304, bhQ, gh, 6912, nullptr, 0, nullptr, 0, 0, 6912);
    k_gru_combine<<<288, 256, 0, stream>>>(gi, gh, hQf, hQf, catQ6, 2368, nullptr, 0);

    // GRU_Sigma
    gemm(catQ6, 2368, WiSigb, 2368, biSig, gi, 6912, nullptr, 0, nullptr, 0, 0, 6912);
    gemm(hSigb, 2304, WhSigb, 2304, bhSig, gh, 6912, nullptr, 0, nullptr, 0, 0, 6912);
    k_gru_combine<<<288, 256, 0, stream>>>(gi, gh, hSigf, nullptr,
                                           catSigS, 4608, catSg3, 4608);

    // o1 = relu(hSig_n @ W1.T + b1) -> catS_in[:, :2304]
    gemm(catSigS, 4608, W1b, 2304, b1, nullptr, 0, catSin, 2400, nullptr, 0, 1, 2304);

    // GRU_S
    gemm(catSin, 2400, WiSb, 2400, biS, gi, 6912, nullptr, 0, nullptr, 0, 0, 6912);
    gemm(catSo2, 4608, WhSb, 2304, bhS, gh, 6912, nullptr, 0, nullptr, 0, 0, 6912);
    k_gru_combine<<<288, 256, 0, stream>>>(gi, gh, hSf, hSf,
                                           catSigS + 2304, 4608, catSo2, 4608);

    // FC2: o2 = relu(cat @ W2a + b2a) @ W2b + b2b
    gemm(catSigS, 4608, W2ab, 4608, b2a, nullptr, 0, o2ab, 4608, nullptr, 0, 1, 4608);
    gemm(o2ab,    4608, W2bb, 4608, b2b, o2f, 2304, catSo2 + 2304, 4608, nullptr, 0, 0, 2304);
    // o3 = relu([hS_n|o2] @ W3 + b3)
    gemm(catSo2, 4608, W3b, 4608, b3, nullptr, 0, catSg3 + 2304, 4608, nullptr, 0, 1, 2304);
    // o4 = relu([hSig_n|o3] @ W4 + b4) -> becomes hSigma carry
    gemm(catSg3, 4608, W4b, 4608, b4, hSigf, 2304, hSigb, 2304, nullptr, 0, 1, 2304);

    k_small_post<<<32, 64, 0, stream>>>(o2f, xprior, dyb, yt,
                                        xpost, xpostp, xpriorp, yprev,
                                        out + (size_t)t * 32 * 48);
  }
}
